// HGCNLP_79044578116123
// MI455X (gfx1250) — compile-verified, matrix operand order checked by reading, not run on hardware
//
#include <hip/hip_runtime.h>
#include <hip/hip_bf16.h>

// ---------------------------------------------------------------------------
// HGCN (3 hyperbolic GCN layers) for MI455X / gfx1250 (wave32, WMMA).
// GEMM: v_wmma_f32_16x16x32_f16, LDS tiles staged with async global->LDS
// (ASYNCcnt path) when the toolchain exposes the builtin. W is converted to
// f16 *transposed* once per layer so both A and B tiles are contiguous 16B
// copies (no scalar ds_store_b16 transpose). Everything else is fused
// row-scaling kernels + an atomic-based SpMM (the memory-bound critical path).
// ---------------------------------------------------------------------------

typedef __attribute__((ext_vector_type(16))) _Float16 v16h;
typedef __attribute__((ext_vector_type(8)))  _Float16 v8h;
typedef __attribute__((ext_vector_type(8)))  float    v8f;
typedef int vi4 __attribute__((vector_size(4 * sizeof(int))));

#define EPSN      1e-7f
#define BALL_EPS  1e-5f

#define DFEAT 1433
#define KP    1440    // K (=D) padded to multiple of BK=32
#define NPAD  1536    // W rows (=output cols) padded to multiple of BN=128
#define BM    128
#define BN    128
#define BK    32

#define AS1 __attribute__((address_space(1)))
#define AS3 __attribute__((address_space(3)))

#if defined(__HIP_DEVICE_COMPILE__) && \
    __has_builtin(__builtin_amdgcn_global_load_async_to_lds_b128)
#define ASYNC_LDS 1
#else
#define ASYNC_LDS 0
#endif

// 16B global -> LDS copy; async (ASYNCcnt) when available.
__device__ __forceinline__ void copy_b128_g2l(void* lds, const void* gmem) {
#if ASYNC_LDS
    __builtin_amdgcn_global_load_async_to_lds_b128(
        (AS1 vi4*)(uintptr_t)gmem, (AS3 vi4*)(uintptr_t)lds, 0, 0);
#else
    *(v8h*)lds = *(const v8h*)gmem;
#endif
}

__device__ __forceinline__ void wait_async_copies() {
#if ASYNC_LDS
#if __has_builtin(__builtin_amdgcn_s_wait_asynccnt)
    __builtin_amdgcn_s_wait_asynccnt(0);
#else
    asm volatile("s_wait_asynccnt 0x0" ::: "memory");
#endif
#endif
}

// ---------------- block-wide sum reduction (wave32 shfl + LDS) -------------
__device__ __forceinline__ float block_reduce_sum(float v, float* scratch) {
    const int lane = threadIdx.x & 31;
    const int wv   = threadIdx.x >> 5;
#pragma unroll
    for (int off = 16; off > 0; off >>= 1) v += __shfl_xor(v, off, 32);
    if (lane == 0) scratch[wv] = v;
    __syncthreads();
    if (threadIdx.x == 0) {
        float t = 0.f;
        const int nw = blockDim.x >> 5;
        for (int i = 0; i < nw; ++i) t += scratch[i];
        scratch[0] = t;
    }
    __syncthreads();
    const float r = scratch[0];
    __syncthreads();            // make scratch reusable by caller
    return r;
}

// ---------------- h0 = expmap0(x, c)  (one block per row) ------------------
__global__ __launch_bounds__(256)
void expmap0_init(const float* __restrict__ x, float* __restrict__ h,
                  int N, int D, float c) {
    __shared__ float scratch[8];
    const int row = blockIdx.x;
    const float* xr = x + (size_t)row * D;
    float*       hr = h + (size_t)row * D;
    float s2 = 0.f;
    for (int j = threadIdx.x; j < D; j += blockDim.x) { float v = xr[j]; s2 += v * v; }
    const float n2 = block_reduce_sum(s2, scratch);
    const float sc = sqrtf(c);
    const float n  = fmaxf(sqrtf(n2), EPSN);
    const float sl = tanhf(sc * n) / (sc * n);
    for (int j = threadIdx.x; j < D; j += blockDim.x) hr[j] = sl * xr[j];
}

// ---- t = logmap0(h, c) written as f16, K-padded; also tn[row]=||t|| -------
__global__ __launch_bounds__(256)
void logmap0_to_f16(const float* __restrict__ h, _Float16* __restrict__ t16,
                    float* __restrict__ tn, int N, int D, float c) {
    __shared__ float scratch[8];
    const int row = blockIdx.x;
    const float* hr = h + (size_t)row * D;
    _Float16*    tr = t16 + (size_t)row * KP;
    float s2 = 0.f;
    for (int j = threadIdx.x; j < D; j += blockDim.x) { float v = hr[j]; s2 += v * v; }
    const float n2 = block_reduce_sum(s2, scratch);
    const float sc = sqrtf(c);
    const float n  = fmaxf(sqrtf(n2), EPSN);
    const float sl = atanhf(fminf(sc * n, 1.f - BALL_EPS)) / (sc * n);
    for (int j = threadIdx.x; j < KP; j += blockDim.x)
        tr[j] = (j < D) ? (_Float16)(sl * hr[j]) : (_Float16)0.f;
    if (threadIdx.x == 0) tn[row] = sl * n;   // uniform scale => ||t|| = sl*n
}

// ---------------- zero the padding rows of t16 (once per launch) -----------
__global__ __launch_bounds__(256)
void zero_t16_pad(_Float16* __restrict__ t16, int startRow, int endRow) {
    const size_t total = (size_t)(endRow - startRow) * KP;
    const size_t idx   = (size_t)blockIdx.x * blockDim.x + threadIdx.x;
    if (idx < total) t16[(size_t)startRow * KP + idx] = (_Float16)0.f;
}

// ------- W fp32 [D x D] -> f16 TRANSPOSED, padded: Wt[n][k], [NPAD][KP] ----
// LDS-tiled 32x32 transpose: coalesced reads along n, coalesced writes along k.
__global__ __launch_bounds__(256)
void convert_w_t(const float* __restrict__ W, _Float16* __restrict__ Wt, int D) {
    __shared__ float tile[32][33];
    const int tk0 = blockIdx.x * 32;      // k tile base (0..KP-32)
    const int tn0 = blockIdx.y * 32;      // n tile base (0..NPAD-32)
    const int lx  = threadIdx.x & 31;
    const int ly  = threadIdx.x >> 5;     // 0..7
#pragma unroll
    for (int i = 0; i < 4; ++i) {
        const int k = tk0 + ly + 8 * i;
        const int n = tn0 + lx;
        tile[ly + 8 * i][lx] = (k < D && n < D) ? W[(size_t)k * D + n] : 0.f;
    }
    __syncthreads();
#pragma unroll
    for (int i = 0; i < 4; ++i) {
        const int n = tn0 + ly + 8 * i;
        const int k = tk0 + lx;
        Wt[(size_t)n * KP + k] = (_Float16)tile[lx][ly + 8 * i];
    }
}

// ---------------- C[N x D] = A[NR x KP](f16) @ Wt[NPAD x KP](f16)^T --------
// 256 threads = 8 waves. Block tile 128x128, K step 32.
// Waves: wm = wave&3 (32-row strip), wn = wave>>2 (64-col strip);
// each wave owns 2x4 = 8 v_wmma_f32_16x16x32_f16 accumulators.
// A rows are padded (zeroed) to a multiple of BM so tile loads are unguarded.
__global__ __launch_bounds__(256)
void wmma_gemm(const _Float16* __restrict__ A, const _Float16* __restrict__ Bt,
               float* __restrict__ C, int N, int Dcols) {
    __shared__ __align__(16) _Float16 As[BM][BK];   // [row][k]
    __shared__ __align__(16) _Float16 Bs[BN][BK];   // [col][k]

    const int tid  = threadIdx.x;
    const int lane = tid & 31;
    const int wave = tid >> 5;
    const int wm   = wave & 3;       // 0..3 -> 32-row strip
    const int wn   = wave >> 2;      // 0..1 -> 64-col strip
    const int rowBase = blockIdx.x * BM;
    const int colBase = blockIdx.y * BN;

    v8f acc[2][4];
#pragma unroll
    for (int i = 0; i < 2; ++i)
#pragma unroll
        for (int j = 0; j < 4; ++j)
#pragma unroll
            for (int e = 0; e < 8; ++e) acc[i][j][e] = 0.f;

    const int r  = tid >> 2;          // 0..63
    const int kc = (tid & 3) * 8;     // 0,8,16,24

    for (int k0 = 0; k0 < KP; k0 += BK) {
        // --- stage A and B tiles (each 128x32 f16) as 16B async copies -----
#pragma unroll
        for (int hlf = 0; hlf < 2; ++hlf) {
            const int row = r + hlf * 64;
            copy_b128_g2l(&As[row][kc],
                          A + (size_t)(rowBase + row) * KP + k0 + kc);
        }
#pragma unroll
        for (int hlf = 0; hlf < 2; ++hlf) {
            const int col = r + hlf * 64;
            copy_b128_g2l(&Bs[col][kc],
                          Bt + (size_t)(colBase + col) * KP + k0 + kc);
        }
        wait_async_copies();
        __syncthreads();

        // --- build fragments per ISA 7.12.2 wave32 layout ------------------
        // lane<16: M/N = lane,     K = {0..7, 16..23}
        // lane>=16: M/N = lane-16, K = {8..15, 24..31}
        const int fr = lane & 15;
        const int kb = (lane < 16) ? 0 : 8;

        v16h afrag[2];
#pragma unroll
        for (int mt = 0; mt < 2; ++mt) {
            const _Float16* p = &As[wm * 32 + mt * 16 + fr][kb];
            const v8h lo = *(const v8h*)p;
            const v8h hi = *(const v8h*)(p + 16);
#pragma unroll
            for (int i = 0; i < 8; ++i) { afrag[mt][i] = lo[i]; afrag[mt][8 + i] = hi[i]; }
        }
        v16h bfrag[4];
#pragma unroll
        for (int nt = 0; nt < 4; ++nt) {
            const _Float16* p = &Bs[wn * 64 + nt * 16 + fr][kb];
            const v8h lo = *(const v8h*)p;
            const v8h hi = *(const v8h*)(p + 16);
#pragma unroll
            for (int i = 0; i < 8; ++i) { bfrag[nt][i] = lo[i]; bfrag[nt][8 + i] = hi[i]; }
        }

#pragma unroll
        for (int mt = 0; mt < 2; ++mt)
#pragma unroll
            for (int nt = 0; nt < 4; ++nt)
                acc[mt][nt] = __builtin_amdgcn_wmma_f32_16x16x32_f16(
                    /*neg_a=*/false, afrag[mt], /*neg_b=*/false, bfrag[nt],
                    /*c_mod=*/(short)0, acc[mt][nt],
                    /*reuse_a=*/false, /*reuse_b=*/false);

        __syncthreads();
    }

    // --- store C: 16x16 f32 layout: lane 0-15 -> M=v, lane 16-31 -> M=8+v --
#pragma unroll
    for (int mt = 0; mt < 2; ++mt) {
#pragma unroll
        for (int nt = 0; nt < 4; ++nt) {
            const int col = colBase + wn * 64 + nt * 16 + (lane & 15);
            if (col >= Dcols) continue;
            const int rBase = rowBase + wm * 32 + mt * 16 + ((lane >= 16) ? 8 : 0);
#pragma unroll
            for (int v = 0; v < 8; ++v) {
                const int row = rBase + v;
                if (row < N) C[(size_t)row * Dcols + col] = acc[mt][nt][v];
            }
        }
    }
}

// ---- h = proj(mobius scale of mx, c_lin) in place; zero aggregation s -----
__global__ __launch_bounds__(256)
void mobius_post(float* __restrict__ mx, const float* __restrict__ tn,
                 float* __restrict__ s_out, int N, int D, float c_lin) {
    __shared__ float scratch[8];
    const int row = blockIdx.x;
    float* mr = mx + (size_t)row * D;
    float* sr = s_out + (size_t)row * D;
    float s2 = 0.f;
    for (int j = threadIdx.x; j < D; j += blockDim.x) { float v = mr[j]; s2 += v * v; }
    const float n2  = block_reduce_sum(s2, scratch);
    const float sc  = sqrtf(c_lin);
    const float mxn = fmaxf(sqrtf(n2), EPSN);
    const float xn  = fmaxf(tn[row], EPSN);
    const float tf  = tanhf((mxn / xn) * atanhf(fminf(sc * xn, 1.f - BALL_EPS)));
    float scale     = tf / (sc * mxn);
    const float nh  = tf / sc;                       // ||result|| (uniform scale)
    const float mxr = (1.f - BALL_EPS) / sc;
    if (nh > mxr) scale *= mxr / nh;                 // proj clip
    for (int j = threadIdx.x; j < D; j += blockDim.x) {
        mr[j] *= scale;
        sr[j]  = 0.f;                                // clear aggregation buffer
    }
}

// ---------------- s[rows[e], :] += vals[e] * h[cols[e], :]  (wave/edge) ----
__global__ __launch_bounds__(256)
void spmm_edges(const int* __restrict__ rows, const int* __restrict__ cols,
                const float* __restrict__ vals, const float* __restrict__ h,
                float* __restrict__ s, int E, int D) {
    const int edge = blockIdx.x * (blockDim.x >> 5) + (threadIdx.x >> 5);
    if (edge >= E) return;
    const int lane = threadIdx.x & 31;
    const int r = rows[edge];
    const int c = cols[edge];
    const float v = vals[edge];
    const float* src = h + (size_t)c * D;
    float*       dst = s + (size_t)r * D;
    for (int j = lane; j < D; j += 32) {
        if (j + 32 < D) __builtin_prefetch(&src[j + 32], 0, 0);
        atomicAdd(&dst[j], v * src[j]);
    }
}

// ---- fused tail: proj(expmap0(relu(logmap0(proj(expmap0(s,c)),c)),c),c) ---
// relu(k*x) = k*relu(x) for k>0 collapses the whole chain to ONE reduction
// pass (two simultaneous sums) + ONE scale pass over the LDS-staged row.
__global__ __launch_bounds__(256)
void post_agg(const float* __restrict__ s, float* __restrict__ out,
              int N, int D, float c) {
    __shared__ float scratch[8];
    __shared__ float rowbuf[NPAD];
    const int row = blockIdx.x;
    const float* sr = s + (size_t)row * D;
    float*       op = out + (size_t)row * D;

    float s2 = 0.f, r2 = 0.f;
    for (int j = threadIdx.x; j < D; j += blockDim.x) {
        const float v = sr[j];
        rowbuf[j] = v;
        s2 += v * v;
        const float rv = fmaxf(v, 0.f);
        r2 += rv * rv;
    }
    const float ns2 = block_reduce_sum(s2, scratch);
    const float nr2 = block_reduce_sum(r2, scratch);
    __syncthreads();

    const float sc   = sqrtf(c);
    const float mxr  = (1.f - BALL_EPS) / sc;
    // expmap0(s, c)
    const float ns   = fmaxf(sqrtf(ns2), EPSN);
    const float sE   = tanhf(sc * ns) / (sc * ns);
    float nE         = tanhf(sc * ns) / sc;
    // proj
    const float sP   = (nE > mxr) ? (mxr / nE) : 1.f;
    nE *= sP;
    // logmap0(agg, c)
    const float nAgg = fmaxf(nE, EPSN);
    const float sL   = atanhf(fminf(sc * nAgg, 1.f - BALL_EPS)) / (sc * nAgg);
    // relu: a_j = S1 * relu(s_j)  with S1 = sE*sP*sL > 0
    const float S1   = sE * sP * sL;
    const float na   = fmaxf(S1 * sqrtf(nr2), EPSN);
    // expmap0(a, c) + proj
    const float sE2  = tanhf(sc * na) / (sc * na);
    float nO         = tanhf(sc * na) / sc;
    const float sP2  = (nO > mxr) ? (mxr / nO) : 1.f;
    const float S2   = sE2 * sP2 * S1;

    for (int j = threadIdx.x; j < D; j += blockDim.x)
        op[j] = S2 * fmaxf(rowbuf[j], 0.f);
}

// ---------------------------------------------------------------------------
extern "C" void kernel_launch(void* const* d_in, const int* in_sizes, int n_in,
                              void* d_out, int out_size, void* d_ws, size_t ws_size,
                              hipStream_t stream) {
    const float* x    = (const float*)d_in[0];
    const float* W[3] = {(const float*)d_in[1], (const float*)d_in[2],
                         (const float*)d_in[3]};
    const float* vals = (const float*)d_in[4];
    const int*   rows = (const int*)d_in[5];
    const int*   cols = (const int*)d_in[6];

    const int D  = DFEAT;
    const int N  = in_sizes[0] / D;
    const int E  = in_sizes[4];
    const int NR = ((N + BM - 1) / BM) * BM;   // A rows padded to tile multiple
    const float c = 0.4f, c_lin = 1.0f;

    // workspace carve (256B aligned)
    char* p = (char*)d_ws;
    auto carve = [&](size_t bytes) -> void* {
        void* q = (void*)p;
        p += (bytes + 255) & ~(size_t)255;
        return q;
    };
    _Float16* t16  = (_Float16*)carve((size_t)NR * KP * sizeof(_Float16));
    _Float16* w16t = (_Float16*)carve((size_t)NPAD * KP * sizeof(_Float16));
    float*    mx   = (float*)carve((size_t)N * D * sizeof(float));
    float*    sbuf = (float*)carve((size_t)N * D * sizeof(float));
    float*    hbuf = (float*)carve((size_t)N * D * sizeof(float));
    float*    tn   = (float*)carve((size_t)N * sizeof(float));
    (void)ws_size;

    const dim3 blk(256);
    const dim3 gemmGrid((N + BM - 1) / BM, (D + BN - 1) / BN);
    const dim3 cwGrid(KP / 32, NPAD / 32);
    const int  eGrid = (E + 7) / 8;   // 8 waves per 256-thread block

    // zero A padding rows once; they are never rewritten
    if (NR > N) {
        const size_t padElems = (size_t)(NR - N) * KP;
        zero_t16_pad<<<(int)((padElems + 255) / 256), blk, 0, stream>>>(t16, N, NR);
    }

    expmap0_init<<<N, blk, 0, stream>>>(x, hbuf, N, D, c);

    for (int l = 0; l < 3; ++l) {
        convert_w_t<<<cwGrid, blk, 0, stream>>>(W[l], w16t, D);
        logmap0_to_f16<<<N, blk, 0, stream>>>(hbuf, t16, tn, N, D, c);
        wmma_gemm<<<gemmGrid, blk, 0, stream>>>(t16, w16t, mx, N, D);
        mobius_post<<<N, blk, 0, stream>>>(mx, tn, sbuf, N, D, c_lin);
        spmm_edges<<<eGrid, blk, 0, stream>>>(rows, cols, vals, mx, sbuf, E, D);
        float* dst = (l == 2) ? (float*)d_out : hbuf;
        post_agg<<<N, blk, 0, stream>>>(sbuf, dst, N, D, c);
    }
}
